// SummaRunner_85718957293693
// MI455X (gfx1250) — matrix operand order repaired and checked by reading, not compiled, not run on hardware
//
#include <hip/hip_runtime.h>
#include <math.h>

// Problem constants (match reference)
#define Bsz 2048
#define Lsz 100
#define Dsz 256
#define TB  16      // batch-rows per workgroup tile

typedef __attribute__((ext_vector_type(16))) __bf16 v16bf;
typedef __attribute__((ext_vector_type(2)))  __bf16 bf2;
typedef __attribute__((ext_vector_type(8)))  float  v8f;

union BF16Frag {
    v16bf v;
    bf2   p2[8];
    uint4 q[2];
};

// ---- conversions / transcendentals: prefer hardware paths ----
#if __has_builtin(__builtin_amdgcn_cvt_pk_bf16_f32)
__device__ __forceinline__ bf2 cvt_pk_bf16(float lo, float hi) {
    return __builtin_amdgcn_cvt_pk_bf16_f32(lo, hi);
}
#else
// plain casts: let the backend select v_cvt_pk_bf16_f32 if the target has it
__device__ __forceinline__ bf2 cvt_pk_bf16(float lo, float hi) {
    bf2 r; r.x = (__bf16)lo; r.y = (__bf16)hi; return r;
}
#endif

#if __has_builtin(__builtin_amdgcn_tanhf)
#define TANHF __builtin_amdgcn_tanhf
#else
#define TANHF tanhf
#endif

__device__ __forceinline__ float fast_sigmoid(float x) {
#if __has_builtin(__builtin_amdgcn_exp2f) && __has_builtin(__builtin_amdgcn_rcpf)
    return __builtin_amdgcn_rcpf(1.0f + __builtin_amdgcn_exp2f(-1.44269504f * x));
#else
    return 1.0f / (1.0f + expf(-x));
#endif
}

// convert 8 consecutive floats (two float4) into 8 packed bf16
__device__ __forceinline__ void pack8(const float4* p, bf2* d) {
    float4 a = p[0], b = p[1];
    d[0] = cvt_pk_bf16(a.x, a.y);
    d[1] = cvt_pk_bf16(a.z, a.w);
    d[2] = cvt_pk_bf16(b.x, b.y);
    d[3] = cvt_pk_bf16(b.z, b.w);
}

// ---------------- tiny precompute kernels ----------------

// one-time f32 -> bf16 conversion of a weight matrix (pairs)
__global__ void k_cvt(const float* __restrict__ src, __bf16* __restrict__ dst, int npairs) {
    int i = blockIdx.x * blockDim.x + threadIdx.x;
    if (i < npairs) {
        float2 f = ((const float2*)src)[i];
        ((bf2*)dst)[i] = cvt_pk_bf16(f.x, f.y);
    }
}

__global__ void k_tables(const float* __restrict__ abs_e, const float* __restrict__ w_abs,
                         const float* __restrict__ rel_e, const float* __restrict__ w_rel,
                         float* __restrict__ absw, float* __restrict__ relw) {
    int t = threadIdx.x;
    if (t < 100) {
        float a = 0.f;
        for (int i = 0; i < 50; ++i) a += abs_e[t*50 + i] * w_abs[i];
        absw[t] = a;
    } else if (t < 125) {
        int s = t - 100; float a = 0.f;
        for (int i = 0; i < 50; ++i) a += rel_e[s*50 + i] * w_rel[i];
        relw[s] = a;
    }
}

__global__ void k_offsets(const int* __restrict__ dl, int* __restrict__ offs) {
    int b = blockIdx.x * blockDim.x + threadIdx.x;
    if (b < Bsz) {
        int s = 0;
        for (int i = 0; i < b; ++i) s += dl[i];
        offs[b] = s;
    }
}

__global__ void k_docs(const float* __restrict__ sent, const int* __restrict__ dl,
                       float* __restrict__ docs) {
    int b = blockIdx.x, d = threadIdx.x;
    int n = dl[b];
    const float* p = sent + (size_t)b * Lsz * Dsz + d;
    float a = 0.f;
    for (int l = 0; l < n; ++l) a += p[(size_t)l * Dsz];
    docs[(size_t)b * Dsz + d] = a / (float)n;
}

// ---------------- WMMA GEMM: out[b,n] = act( sum_k A[b,k]*W[n,k] + bias[n] ) ----------------
// A: [Bsz,256] f32; Wb: [256,256] bf16 row-major (row = n, consecutive k).
__global__ void __launch_bounds__(256) k_gemm(const float* __restrict__ A,
                                              const __bf16* __restrict__ Wb,
                                              const float* __restrict__ bias,
                                              float* __restrict__ outp, int act_tanh) {
    __shared__ __bf16 tA[TB][Dsz];
    int tid = threadIdx.x;
    int b0  = blockIdx.x * TB;

    { // stage A -> bf16 LDS
        int br = tid >> 4, dg = tid & 15;
        const float4* src = (const float4*)&A[(size_t)(b0 + br) * Dsz + dg * 16];
        bf2* dst = (bf2*)&tA[br][dg * 16];
        pack8(src, dst);
        pack8(src + 2, dst + 4);
    }
    __syncthreads();

    int w = tid >> 5, lane = tid & 31;
    int m = lane & 15, hi = lane >> 4;
    int col0 = w * 32 + m, col1 = col0 + 16;
    v8f acc0 = {}; v8f acc1 = {};

#pragma unroll
    for (int k = 0; k < 8; ++k) {
        BF16Frag Af;
        Af.q[0] = *(const uint4*)&tA[m][k * 32 + hi * 16];
        Af.q[1] = *(const uint4*)&tA[m][k * 32 + hi * 16 + 8];
        int e0 = k * 32 + hi * 16;
        BF16Frag Bf0, Bf1;
        const uint4* p0 = (const uint4*)&Wb[(size_t)col0 * Dsz + e0];
        const uint4* p1 = (const uint4*)&Wb[(size_t)col1 * Dsz + e0];
        Bf0.q[0] = p0[0]; Bf0.q[1] = p0[1];
        Bf1.q[0] = p1[0]; Bf1.q[1] = p1[1];
        acc0 = __builtin_amdgcn_wmma_f32_16x16x32_bf16(false, Af.v, false, Bf0.v, (short)0, acc0, false, false);
        acc1 = __builtin_amdgcn_wmma_f32_16x16x32_bf16(false, Af.v, false, Bf1.v, (short)0, acc1, false, false);
    }

    float bia0 = bias ? bias[col0] : 0.f;
    float bia1 = bias ? bias[col1] : 0.f;
#pragma unroll
    for (int j = 0; j < 8; ++j) {
        int row = hi * 8 + j;
        float v0 = acc0[j] + bia0;
        float v1 = acc1[j] + bia1;
        if (act_tanh) { v0 = TANHF(v0); v1 = TANHF(v1); }
        outp[(size_t)(b0 + row) * Dsz + col0] = v0;
        outp[(size_t)(b0 + row) * Dsz + col1] = v1;
    }
}

// ---------------- pre[b,l] = sent[b,l,:]·(u[b,:]+w_content) + absw[l] + relw[idx] + bias ----
__global__ void __launch_bounds__(256) k_pre(const float* __restrict__ sent,
                                             const float* __restrict__ u,
                                             const float* __restrict__ wcont,
                                             const float* __restrict__ absw,
                                             const float* __restrict__ relw,
                                             const float* __restrict__ biasp,
                                             const int* __restrict__ dl,
                                             float* __restrict__ pre) {
    int b = blockIdx.x;
    int lane = threadIdx.x & 31, w = threadIdx.x >> 5;
    float uc[8];
#pragma unroll
    for (int j = 0; j < 8; ++j) {
        int d = lane + 32 * j;
        uc[j] = u[(size_t)b * Dsz + d] + wcont[d];
    }
    float dlf  = (float)dl[b];
    float bias = biasp[0];
    for (int l = w; l < Lsz; l += 8) {
        const float* sp = &sent[((size_t)b * Lsz + l) * Dsz];
        float acc = 0.f;
#pragma unroll
        for (int j = 0; j < 8; ++j) acc += sp[lane + 32 * j] * uc[j];
#pragma unroll
        for (int off = 16; off >= 1; off >>= 1) acc += __shfl_xor(acc, off, 32);
        if (lane == 0) {
            int ridx = (int)rintf((float)(l + 1) * 9.0f / dlf);
            ridx = ridx < 0 ? 0 : (ridx > 24 ? 24 : ridx);
            pre[(size_t)b * Lsz + l] = acc + absw[l] + relw[ridx] + bias;
        }
    }
}

// ---------------- recurrence: s_{t+1} = s_t + p_t*h_t, p = sigmoid(pre - h·W_nov·tanh(s)) ----
__global__ void __launch_bounds__(256) k_recur(const float* __restrict__ sent,
                                               const __bf16* __restrict__ Wnovb,
                                               const float* __restrict__ pre,
                                               const int* __restrict__ dl,
                                               const int* __restrict__ offs,
                                               float* __restrict__ outp) {
    __shared__ __bf16 tsA[TB][Dsz];           // bf16 tanh(s) tile (A matrix)      8 KB
    __shared__ float  sst[TB][Dsz];           // f32 state s                      16 KB
    __shared__ float  mld[TB][Dsz];           // f32 m = tanh(s)@Wnovᵀ            16 KB

    int tid = threadIdx.x;
    int b0  = blockIdx.x * TB;
    int w = tid >> 5, lane = tid & 31;
    int m = lane & 15, hi = lane >> 4;

    // Preload B fragments: B[e,d] = W_nov[d,e]; lane holds column d, 16 consecutive e.
    BF16Frag Bf[8][2];
    {
        int c0 = w * 32 + m;
#pragma unroll
        for (int k = 0; k < 8; ++k) {
            int e0 = k * 32 + hi * 16;
            const uint4* p0 = (const uint4*)&Wnovb[(size_t)c0 * Dsz + e0];
            const uint4* p1 = (const uint4*)&Wnovb[(size_t)(c0 + 16) * Dsz + e0];
            Bf[k][0].q[0] = p0[0]; Bf[k][0].q[1] = p0[1];
            Bf[k][1].q[0] = p1[0]; Bf[k][1].q[1] = p1[1];
        }
    }

    // phase-3 identity: thread owns (batch row pb, 16-wide d chunk dg)
    int pb = tid >> 4, dg = tid & 15, d0 = dg * 16;
    int gb = b0 + pb;
    int mydl  = dl[gb];
    int myoff = offs[gb];
    const float* prep = &pre[(size_t)gb * Lsz];
    const float4* hbase = (const float4*)&sent[(size_t)gb * Lsz * Dsz + d0];

#pragma unroll
    for (int j = 0; j < 8; ++j) {
        sst[pb][d0 + 2*j] = 0.f; sst[pb][d0 + 2*j + 1] = 0.f;
        *(bf2*)&tsA[pb][d0 + 2*j] = cvt_pk_bf16(0.f, 0.f);
    }
    __syncthreads();

    for (int t = 0; t < Lsz; ++t) {
        // --- issue h tile + pre loads early; latency hides under the WMMA chain ---
        const float4* hp = hbase + (size_t)t * (Dsz / 4);
        float4 hq0 = hp[0], hq1 = hp[1], hq2 = hp[2], hq3 = hp[3];
        float pre_t = prep[t];

        // --- GEMM: m = tanh(s) @ Wnovᵀ  (M=16, N=256, K=256; 16 WMMAs/wave) ---
        v8f acc0 = {}; v8f acc1 = {};
#pragma unroll
        for (int k = 0; k < 8; ++k) {
            BF16Frag Af;
            Af.q[0] = *(const uint4*)&tsA[m][k * 32 + hi * 16];
            Af.q[1] = *(const uint4*)&tsA[m][k * 32 + hi * 16 + 8];
            acc0 = __builtin_amdgcn_wmma_f32_16x16x32_bf16(false, Af.v, false, Bf[k][0].v, (short)0, acc0, false, false);
            acc1 = __builtin_amdgcn_wmma_f32_16x16x32_bf16(false, Af.v, false, Bf[k][1].v, (short)0, acc1, false, false);
        }
#pragma unroll
        for (int j = 0; j < 8; ++j) {
            int row = hi * 8 + j;
            mld[row][w * 32 + m]      = acc0[j];
            mld[row][w * 32 + m + 16] = acc1[j];
        }
        __syncthreads();

        // --- nov dot, sigmoid, state update ---
        float h[16];
        h[0]=hq0.x; h[1]=hq0.y; h[2]=hq0.z; h[3]=hq0.w;
        h[4]=hq1.x; h[5]=hq1.y; h[6]=hq1.z; h[7]=hq1.w;
        h[8]=hq2.x; h[9]=hq2.y; h[10]=hq2.z; h[11]=hq2.w;
        h[12]=hq3.x; h[13]=hq3.y; h[14]=hq3.z; h[15]=hq3.w;
        float part = 0.f;
#pragma unroll
        for (int i = 0; i < 16; ++i) part += h[i] * mld[pb][d0 + i];
#pragma unroll
        for (int off = 8; off >= 1; off >>= 1) part += __shfl_xor(part, off, 16);
        float p = fast_sigmoid(pre_t - part);     // nov = -part
#pragma unroll
        for (int j = 0; j < 8; ++j) {
            float n0 = sst[pb][d0 + 2*j]     + p * h[2*j];
            float n1 = sst[pb][d0 + 2*j + 1] + p * h[2*j + 1];
            sst[pb][d0 + 2*j]     = n0;
            sst[pb][d0 + 2*j + 1] = n1;
            *(bf2*)&tsA[pb][d0 + 2*j] = cvt_pk_bf16(TANHF(n0), TANHF(n1));
        }
        if (dg == 0 && t < mydl) outp[myoff + t] = p;
        __syncthreads();   // tsA ready for next step's GEMM
    }
}

// ---------------- launch ----------------
extern "C" void kernel_launch(void* const* d_in, const int* in_sizes, int n_in,
                              void* d_out, int out_size, void* d_ws, size_t ws_size,
                              hipStream_t stream) {
    const float* sent   = (const float*)d_in[0];
    const float* fc_w   = (const float*)d_in[1];
    const float* fc_b   = (const float*)d_in[2];
    const float* w_cont = (const float*)d_in[3];
    const float* W_sal  = (const float*)d_in[4];
    const float* W_nov  = (const float*)d_in[5];
    const float* abs_e  = (const float*)d_in[6];
    const float* rel_e  = (const float*)d_in[7];
    const float* w_abs  = (const float*)d_in[8];
    const float* w_rel  = (const float*)d_in[9];
    const float* bias   = (const float*)d_in[10];
    const int*   dlens  = (const int*)d_in[11];

    // workspace layout (~5.4 MB total)
    char* ws = (char*)d_ws;
    float*  docs_u  = (float*)(ws);                               // 2 MB: docs, later reused as u
    float*  doc     = (float*)(ws + (2u << 20));                  // 2 MB
    float*  pre     = (float*)(ws + (4u << 20));                  // 819200 B
    float*  absw    = (float*)(ws + (4u << 20) + 819200);         // 400 B (pad 512)
    float*  relw    = (float*)(ws + (4u << 20) + 819200 + 512);   // 100 B (pad 512)
    int*    offs    = (int*)  (ws + (4u << 20) + 819200 + 1024);  // 8192 B
    __bf16* fcw_bf  = (__bf16*)(ws + (5u << 20));                 // 128 KB
    __bf16* wsal_bf = (__bf16*)(ws + (5u << 20) + (128u << 10));  // 128 KB
    __bf16* wnov_bf = (__bf16*)(ws + (5u << 20) + (256u << 10));  // 128 KB

    const int NPAIRS = Dsz * Dsz / 2;   // 32768
    k_cvt    <<<NPAIRS / 256, 256, 0, stream>>>(fc_w,  fcw_bf,  NPAIRS);
    k_cvt    <<<NPAIRS / 256, 256, 0, stream>>>(W_sal, wsal_bf, NPAIRS);
    k_cvt    <<<NPAIRS / 256, 256, 0, stream>>>(W_nov, wnov_bf, NPAIRS);
    k_tables <<<1,   128, 0, stream>>>(abs_e, w_abs, rel_e, w_rel, absw, relw);
    k_offsets<<<8,   256, 0, stream>>>(dlens, offs);
    k_docs   <<<Bsz, Dsz, 0, stream>>>(sent, dlens, docs_u);
    k_gemm   <<<Bsz / TB, 256, 0, stream>>>(docs_u, fcw_bf, fc_b, doc, 1);
    k_gemm   <<<Bsz / TB, 256, 0, stream>>>(doc, wsal_bf, nullptr, docs_u, 0);
    k_pre    <<<Bsz, 256, 0, stream>>>(sent, docs_u, w_cont, absw, relw, bias, dlens, pre);
    k_recur  <<<Bsz / TB, 256, 0, stream>>>(sent, wnov_bf, pre, dlens, offs, (float*)d_out);
}
